// Mapper_53446573032155
// MI455X (gfx1250) — matrix operand order repaired and checked by reading, not compiled, not run on hardware
//
#include <hip/hip_runtime.h>

// ---------- types ----------
typedef __bf16 v16bf __attribute__((ext_vector_type(16)));
typedef __bf16 v2bf  __attribute__((ext_vector_type(2)));
typedef float  v8f   __attribute__((ext_vector_type(8)));
typedef unsigned short v8us  __attribute__((ext_vector_type(8)));
typedef unsigned short v16us __attribute__((ext_vector_type(16)));

#define WN    20     // number of words
#define BATCH 256
#define MT    256    // M tile = full batch -> each weight element read exactly once
#define NT    128    // N tile
#define KT    32     // K tile (one wmma K-step)
#define LDA   40     // padded LDS stride (elements) for A tile  (32 used + 8 pad)
#define LDB   40     // padded LDS stride (elements) for B tile

// fp32x2 -> packed bf16x2 via hardware v_cvt_pk_bf16_f32 (RNE)
__device__ __forceinline__ unsigned pack_bf16(float lo, float hi) {
#if __has_builtin(__builtin_amdgcn_cvt_pk_bf16_f32)
    return __builtin_bit_cast(unsigned, __builtin_amdgcn_cvt_pk_bf16_f32(lo, hi));
#else
    v2bf v = { (__bf16)lo, (__bf16)hi };
    return __builtin_bit_cast(unsigned, v);
#endif
}

__device__ __forceinline__ v16bf load_frag_a(const unsigned short* p) {
    // interleaved A layout: 8 contiguous bf16, then 8 more at +16 elements
    v8us lo = *(const v8us*)p;
    v8us hi = *(const v8us*)(p + 16);
    v16us m = __builtin_shufflevector(lo, hi, 0,1,2,3,4,5,6,7,8,9,10,11,12,13,14,15);
    return __builtin_bit_cast(v16bf, m);
}

__device__ __forceinline__ v16bf load_frag_b(const unsigned short* p) {
    // interleaved layout: 8 contiguous bf16, then 8 more at +16 elements
    v8us lo = *(const v8us*)p;
    v8us hi = *(const v8us*)(p + 16);
    v16us m = __builtin_shufflevector(lo, hi, 0,1,2,3,4,5,6,7,8,9,10,11,12,13,14,15);
    return __builtin_bit_cast(v16bf, m);
}

// A: each thread owns one row (tid) and all 32 k's; B: k-pair x 8 n's (transpose-store)
__device__ __forceinline__ void stage_tiles(unsigned short* aDst, unsigned short* bDst,
                                            const float4* ra, const float4* rb,
                                            int ar, int bkp, int bns) {
    unsigned int* da = (unsigned int*)(aDst + ar * LDA);
    const float* fa = (const float*)ra;
#pragma unroll
    for (int j = 0; j < 16; ++j)
        da[j] = pack_bf16(fa[2 * j], fa[2 * j + 1]);
#pragma unroll
    for (int j = 0; j < 8; ++j) {
        float lo = ((const float*)&rb[(j >> 2)])[j & 3];      // row k
        float hi = ((const float*)&rb[2 + (j >> 2)])[j & 3];  // row k+1
        *(unsigned int*)(bDst + (bns + j) * LDB + bkp) = pack_bf16(lo, hi);
    }
}

// C[b,w,n] = sum_k A[b,w,k] * Wg[w,k,n] + bias[w,n]
// A logical layout [B, WN, K]; Wg [WN, K, N]; bias [WN, N]; C [B, WN, N]
__global__ __launch_bounds__(256)
void gemm_bias_kernel(const float* __restrict__ A, const float* __restrict__ Wg,
                      const float* __restrict__ bias, float* __restrict__ C,
                      int K, int N) {
    __shared__ __align__(16) unsigned short aLds[2][MT * LDA];  // 2 x 20 KB
    __shared__ __align__(16) unsigned short bLds[2][NT * LDB];  // 2 x 10 KB

    const int tid  = threadIdx.x;
    const int lane = tid & 31;
    const int wave = tid >> 5;
    const int wm   = wave & 3;      // 4 row-strips of 64
    const int wn   = wave >> 2;     // 2 col-strips of 64
    const int lr   = lane & 15;
    const int kh   = lane >> 4;     // lane half selects K sub-range

    const int w       = blockIdx.z;
    const int colBase = blockIdx.x * NT;
    const int rowBase = blockIdx.y * MT;   // gridDim.y == 1

    // staging assignment
    const int ar  = tid;               // A row (256 rows x 1 thread, 32 k each)
    const int bkp = (tid >> 4) * 2;    // B k-pair
    const int bns = (tid & 15) * 8;    // B n-segment (8 wide)

    const float* aSrc = A  + ((size_t)(rowBase + ar) * WN + w) * K;
    const float* bSrc = Wg + ((size_t)w * K + bkp) * N + colBase + bns;

    float4 ra[8], rb[4];
    {   // prefetch tile 0
        const float4* p = (const float4*)aSrc;
#pragma unroll
        for (int j = 0; j < 8; ++j) ra[j] = p[j];
        const float4* q0 = (const float4*)bSrc;
        const float4* q1 = (const float4*)(bSrc + N);
        rb[0] = q0[0]; rb[1] = q0[1]; rb[2] = q1[0]; rb[3] = q1[1];
    }

    v8f acc[4][4];
#pragma unroll
    for (int mi = 0; mi < 4; ++mi)
#pragma unroll
        for (int ni = 0; ni < 4; ++ni) acc[mi][ni] = (v8f){0.f,0.f,0.f,0.f,0.f,0.f,0.f,0.f};

    const int nK = K / KT;
    stage_tiles(aLds[0], bLds[0], ra, rb, ar, bkp, bns);
    __syncthreads();

    int buf = 0;
    for (int kt = 0; kt < nK; ++kt) {
        const bool hasNext = (kt + 1) < nK;
        if (hasNext) {  // global prefetch for next K tile (overlaps WMMA below)
            const float4* p = (const float4*)(aSrc + (size_t)(kt + 1) * KT);
#pragma unroll
            for (int j = 0; j < 8; ++j) ra[j] = p[j];
            const float*  bn_ = bSrc + (size_t)(kt + 1) * KT * N;
            const float4* q0  = (const float4*)bn_;
            const float4* q1  = (const float4*)(bn_ + N);
            rb[0] = q0[0]; rb[1] = q0[1]; rb[2] = q1[0]; rb[3] = q1[1];
        }

        // ---- compute on current LDS buffer: 16 wmma / iter / wave ----
        v16bf af[4];
#pragma unroll
        for (int mi = 0; mi < 4; ++mi)
            af[mi] = load_frag_a(&aLds[buf][(wm * 64 + mi * 16 + lr) * LDA + kh * 8]);
#pragma unroll
        for (int ni = 0; ni < 4; ++ni) {
            v16bf bf = load_frag_b(&bLds[buf][(wn * 64 + ni * 16 + lr) * LDB + kh * 8]);
#pragma unroll
            for (int mi = 0; mi < 4; ++mi)
                acc[mi][ni] = __builtin_amdgcn_wmma_f32_16x16x32_bf16(
                    false, af[mi], false, bf, (short)0, acc[mi][ni], false, false);
        }

        if (hasNext)
            stage_tiles(aLds[buf ^ 1], bLds[buf ^ 1], ra, rb, ar, bkp, bns);
        __syncthreads();
        buf ^= 1;
    }

    // ---- epilogue: bias + store (C/D layout: row = v + 8*kh, col = lr) ----
#pragma unroll
    for (int ni = 0; ni < 4; ++ni) {
        const int col = colBase + wn * 64 + ni * 16 + lr;
        const float bv = bias[w * N + col];
#pragma unroll
        for (int mi = 0; mi < 4; ++mi) {
            const int r0 = rowBase + wm * 64 + mi * 16 + kh * 8;
#pragma unroll
            for (int v = 0; v < 8; ++v)
                C[((size_t)(r0 + v) * WN + w) * N + col] = acc[mi][ni][v] + bv;
        }
    }
}

// In-place LayerNorm (over H=1280) + LeakyReLU(0.01); one block per (b,w) row.
__global__ __launch_bounds__(256)
void ln_lrelu_kernel(float* __restrict__ h, const float* __restrict__ g,
                     const float* __restrict__ bb) {
    const int H   = 1280;
    const int row = blockIdx.x;       // b*WN + w
    const int w   = row % WN;
    const int tid = threadIdx.x;
    float* x = h + (size_t)row * H;

    float v[5], s = 0.f;
#pragma unroll
    for (int i = 0; i < 5; ++i) { v[i] = x[tid + i * 256]; s += v[i]; }

    __shared__ float red[256];
    red[tid] = s;
    __syncthreads();
    for (int off = 128; off > 0; off >>= 1) {
        if (tid < off) red[tid] += red[tid + off];
        __syncthreads();
    }
    const float mu = red[0] * (1.0f / H);
    __syncthreads();

    float s2 = 0.f;
#pragma unroll
    for (int i = 0; i < 5; ++i) { float d = v[i] - mu; s2 += d * d; }
    red[tid] = s2;
    __syncthreads();
    for (int off = 128; off > 0; off >>= 1) {
        if (tid < off) red[tid] += red[tid + off];
        __syncthreads();
    }
    const float rstd = rsqrtf(red[0] * (1.0f / H) + 1e-5f);

#pragma unroll
    for (int i = 0; i < 5; ++i) {
        const int hh = tid + i * 256;
        float y = (v[i] - mu) * rstd * g[w * H + hh] + bb[w * H + hh];
        x[hh] = (y >= 0.f) ? y : 0.01f * y;
    }
}

extern "C" void kernel_launch(void* const* d_in, const int* in_sizes, int n_in,
                              void* d_out, int out_size, void* d_ws, size_t ws_size,
                              hipStream_t stream) {
    (void)in_sizes; (void)n_in; (void)out_size; (void)ws_size;
    const float* embs = (const float*)d_in[0];
    const float* W1   = (const float*)d_in[1];
    const float* b1   = (const float*)d_in[2];
    const float* g1   = (const float*)d_in[3];
    const float* bn1  = (const float*)d_in[4];
    const float* W2   = (const float*)d_in[5];
    const float* b2   = (const float*)d_in[6];
    const float* g2   = (const float*)d_in[7];
    const float* bn2  = (const float*)d_in[8];
    const float* W3   = (const float*)d_in[9];
    const float* b3   = (const float*)d_in[10];
    const float* g3   = (const float*)d_in[11];
    const float* bn3  = (const float*)d_in[12];
    const float* W4   = (const float*)d_in[13];
    const float* b4   = (const float*)d_in[14];
    float* out = (float*)d_out;

    float* bufA = (float*)d_ws;                          // [256,20,1280]
    float* bufB = bufA + (size_t)BATCH * WN * 1280;      // [256,20,1280]

    const dim3 blk(256);
    const dim3 lnGrid(BATCH * WN);

    // L1: [256,1024] @ [1024,1280]
    gemm_bias_kernel<<<dim3(1280 / NT, 1, WN), blk, 0, stream>>>(embs, W1, b1, bufA, 1024, 1280);
    ln_lrelu_kernel<<<lnGrid, blk, 0, stream>>>(bufA, g1, bn1);
    // L2: [256,1280] @ [1280,1280]
    gemm_bias_kernel<<<dim3(1280 / NT, 1, WN), blk, 0, stream>>>(bufA, W2, b2, bufB, 1280, 1280);
    ln_lrelu_kernel<<<lnGrid, blk, 0, stream>>>(bufB, g2, bn2);
    // L3: [256,1280] @ [1280,1280]
    gemm_bias_kernel<<<dim3(1280 / NT, 1, WN), blk, 0, stream>>>(bufB, W3, b3, bufA, 1280, 1280);
    ln_lrelu_kernel<<<lnGrid, blk, 0, stream>>>(bufA, g3, bn3);
    // L4: [256,1280] @ [1280,1024] -> output
    gemm_bias_kernel<<<dim3(1024 / NT, 1, WN), blk, 0, stream>>>(bufA, W4, b4, out, 1280, 1024);
}
